// MultiHeadAttention_43636867728011
// MI455X (gfx1250) — matrix operand order repaired and checked by reading, not compiled, not run on hardware
//
#include <hip/hip_runtime.h>

// ---------------------------------------------------------------------------
// MHA forward for B=2, T=2048, C=1024, H=16, D=64 on gfx1250 (wave32, WMMA).
// bf16 A/B operands, f32 accumulation via v_wmma_f32_16x16x32_bf16.
// 32-row M-blocking per wave: every B-fragment feeds two WMMAs.
// ---------------------------------------------------------------------------

typedef __attribute__((ext_vector_type(16))) __bf16 v16bf;
typedef __attribute__((ext_vector_type(8)))  float  v8f;

#define EMBED 1024
#define SEQ   2048
#define NH    16
#define HD    64
#define NROW  4096            // B*T

// 1/sqrt(64) * log2(e): scores pre-scaled into log2 domain for exp2f softmax
#define SOFT_SCALE 0.18033688011112042f

__device__ __forceinline__ unsigned short f2bf(float f) {
  union { float f; unsigned u; } c; c.f = f;
  unsigned u = c.u;
  u += 0x7fffu + ((u >> 16) & 1u);   // round-to-nearest-even
  return (unsigned short)(u >> 16);
}

// K offset of A-fragment dword v for 16x16x32 bf16 WMMA (contiguous: k = 16*half + 2v).
__device__ __forceinline__ unsigned a_kk(int v, unsigned half) {
  return (unsigned)(2 * v) + 16u * half;
}

// ---------------------------------------------------------------------------
// Kernel 1: fp32 -> bf16 conversion of x, qkv_w, out_w
// ---------------------------------------------------------------------------
__global__ void cvt_kernel(const float* __restrict__ x,
                           const float* __restrict__ w1,
                           const float* __restrict__ w2,
                           unsigned short* __restrict__ xb,
                           unsigned short* __restrict__ w1b,
                           unsigned short* __restrict__ w2b,
                           int nx, int nw1, int nw2) {
  int i = blockIdx.x * blockDim.x + threadIdx.x;
  int total = nx + nw1 + nw2;
  if (i >= total) return;
  if (i < nx)            xb[i] = f2bf(x[i]);
  else if (i < nx + nw1) w1b[i - nx] = f2bf(w1[i - nx]);
  else                   w2b[i - nx - nw1] = f2bf(w2[i - nx - nw1]);
}

// ---------------------------------------------------------------------------
// Kernel 2: QKV projection. qkv = x @ qkv_w^T + qkv_b, scattered into
//   Q  [B,H,T,D] bf16, K [B,H,T,D] bf16, Vt [B,H,D,T] bf16 (transposed).
// Each wave: 32x64 output block (2 M-tiles x 4 N-tiles), K-loop step 32.
// ---------------------------------------------------------------------------
__global__ __launch_bounds__(256) void qkv_gemm(
    const unsigned short* __restrict__ xb,   // [4096,1024]
    const unsigned short* __restrict__ wb,   // [3072,1024]
    const float* __restrict__ bias,          // [3072]
    unsigned short* __restrict__ qb,
    unsigned short* __restrict__ kb,
    unsigned short* __restrict__ vtb) {
  const unsigned lane  = threadIdx.x & 31u;
  const unsigned wid   = threadIdx.x >> 5;
  const unsigned mlane = lane & 15u;
  const unsigned half  = lane >> 4;
  const unsigned m0 = (blockIdx.x * 8u + wid) * 32u;
  const unsigned n0 = blockIdx.y * 64u;

  v8f acc[2][4] = {{{}, {}, {}, {}}, {{}, {}, {}, {}}};
  const unsigned* x32 = (const unsigned*)xb;

  for (unsigned k0 = 0; k0 < EMBED; k0 += 32) {
    union { v16bf v; unsigned u[8]; } a[2];
#pragma unroll
    for (int i = 0; i < 2; ++i)
#pragma unroll
      for (int v = 0; v < 8; ++v) {
        unsigned kk = a_kk(v, half) + k0;
        a[i].u[v] = x32[((m0 + i * 16u + mlane) * EMBED + kk) >> 1];
      }
#pragma unroll
    for (int t = 0; t < 4; ++t) {
      unsigned n = n0 + t * 16u + mlane;
      const uint4* bp = (const uint4*)(wb + n * EMBED + k0 + 16u * half);
      union { v16bf v; uint4 q[2]; } bm;
      bm.q[0] = bp[0]; bm.q[1] = bp[1];
#pragma unroll
      for (int i = 0; i < 2; ++i)
        acc[i][t] = __builtin_amdgcn_wmma_f32_16x16x32_bf16(
            false, a[i].v, false, bm.v, (short)0, acc[i][t], false, false);
    }
  }

#pragma unroll
  for (int t = 0; t < 4; ++t) {
    unsigned n = n0 + t * 16u + mlane;
    float bv = bias[n];
    unsigned sel = n >> 10;           // 0:q 1:k 2:v
    unsigned c = n & 1023u;
    unsigned h = c >> 6;
    unsigned d = c & 63u;
#pragma unroll
    for (int i = 0; i < 2; ++i)
#pragma unroll
      for (int r = 0; r < 8; ++r) {
        unsigned gm = m0 + i * 16u + r + 8u * half;   // global row [0,4096)
        unsigned bi = gm >> 11;                       // batch
        unsigned tp = gm & 2047u;                     // token
        unsigned short val = f2bf(acc[i][t][r] + bv);
        size_t bh = (size_t)(bi * NH + h);
        if (sel == 0)      qb[(bh * SEQ + tp) * HD + d] = val;
        else if (sel == 1) kb[(bh * SEQ + tp) * HD + d] = val;
        else               vtb[(bh * HD + d) * SEQ + tp] = val;
      }
  }
}

// ---------------------------------------------------------------------------
// Kernel 3: causal flash attention. One wave = one 32-row Q tile of one
// (b,h). Online softmax in log2 domain, 32 keys per step, 16 WMMAs per
// step. P re-layout (C-layout -> A-layout) via per-wave LDS buffer.
// ---------------------------------------------------------------------------
__global__ __launch_bounds__(256) void attn_kernel(
    const unsigned short* __restrict__ qb,
    const unsigned short* __restrict__ kb,
    const unsigned short* __restrict__ vtb,
    unsigned short* __restrict__ attb) {          // [4096,1024] bf16
  __shared__ unsigned short Pbuf[8 * 32 * 32];    // 2KB per wave

  const unsigned lane  = threadIdx.x & 31u;
  const unsigned wid   = threadIdx.x >> 5;
  const unsigned mlane = lane & 15u;
  const unsigned half  = lane >> 4;

  unsigned qt  = blockIdx.x * 8u + wid;     // 0..2047
  unsigned b   = qt >> 10;                  // 16 heads * 64 tiles
  unsigned rem = qt & 1023u;
  unsigned h   = rem >> 6;
  unsigned q0  = (rem & 63u) << 5;          // 32-row tile
  size_t base = (size_t)(b * NH + h) * SEQ * HD;  // same extent for q/k/vt

  // Q A-fragments: [i = row sub-tile][dh = d-half], loaded once
  union { v16bf v; unsigned u[8]; } aq[2][2];
  const unsigned* q32 = (const unsigned*)qb;
#pragma unroll
  for (int i = 0; i < 2; ++i)
#pragma unroll
    for (int v = 0; v < 8; ++v) {
      size_t idx = base + (size_t)(q0 + i * 16u + mlane) * HD + a_kk(v, half);
      aq[i][0].u[v] = q32[idx >> 1];
      aq[i][1].u[v] = q32[(idx + 32) >> 1];
    }

  float mi[2][8], li[2][8];
#pragma unroll
  for (int i = 0; i < 2; ++i)
#pragma unroll
    for (int r = 0; r < 8; ++r) { mi[i][r] = -3.0e38f; li[i][r] = 0.0f; }
  v8f o[2][4] = {{{}, {}, {}, {}}, {{}, {}, {}, {}}};

  unsigned short* pb = &Pbuf[wid * 1024u];
  const unsigned short* kbase = kb + base;
  const unsigned short* vbase = vtb + base;

  for (unsigned jb = 0; jb <= q0 + 31u; jb += 32) {
    // --- scores: 2 row-subtiles x 2 key tiles, sharing K B-fragments ---
    v8f s[2][2];
#pragma unroll
    for (int kt = 0; kt < 2; ++kt) {
      unsigned key = jb + kt * 16u + mlane;
      const uint4* kp0 = (const uint4*)(kbase + (size_t)key * HD + 16u * half);
      const uint4* kp1 = (const uint4*)(kbase + (size_t)key * HD + 32u + 16u * half);
      union { v16bf v; uint4 q[2]; } blo, bhi;
      blo.q[0] = kp0[0]; blo.q[1] = kp0[1];
      bhi.q[0] = kp1[0]; bhi.q[1] = kp1[1];
#pragma unroll
      for (int i = 0; i < 2; ++i) {
        v8f c = {};
        c = __builtin_amdgcn_wmma_f32_16x16x32_bf16(
            false, aq[i][0].v, false, blo.v, (short)0, c, false, false);
        c = __builtin_amdgcn_wmma_f32_16x16x32_bf16(
            false, aq[i][1].v, false, bhi.v, (short)0, c, false, false);
        s[i][kt] = c;
      }
    }

    // --- scale into log2 domain + causal mask ---
#pragma unroll
    for (int i = 0; i < 2; ++i)
#pragma unroll
      for (int kt = 0; kt < 2; ++kt)
#pragma unroll
        for (int r = 0; r < 8; ++r) {
          unsigned qidx = q0 + i * 16u + r + 8u * half;
          unsigned kidx = jb + kt * 16u + mlane;
          float v = s[i][kt][r] * SOFT_SCALE;
          s[i][kt][r] = (kidx <= qidx) ? v : -3.0e38f;
        }

    // --- online softmax (reductions across the 16-lane half-group) ---
#pragma unroll
    for (int i = 0; i < 2; ++i)
#pragma unroll
      for (int r = 0; r < 8; ++r) {
        float mrow = fmaxf(s[i][0][r], s[i][1][r]);
        mrow = fmaxf(mrow, __shfl_xor(mrow, 1, 32));
        mrow = fmaxf(mrow, __shfl_xor(mrow, 2, 32));
        mrow = fmaxf(mrow, __shfl_xor(mrow, 4, 32));
        mrow = fmaxf(mrow, __shfl_xor(mrow, 8, 32));
        float mall = fmaxf(mi[i][r], mrow);
        float al = exp2f(mi[i][r] - mall);
        float p0 = exp2f(s[i][0][r] - mall);
        float p1 = exp2f(s[i][1][r] - mall);
        s[i][0][r] = p0; s[i][1][r] = p1;
        float ps = p0 + p1;
        ps += __shfl_xor(ps, 1, 32);
        ps += __shfl_xor(ps, 2, 32);
        ps += __shfl_xor(ps, 4, 32);
        ps += __shfl_xor(ps, 8, 32);
        li[i][r] = li[i][r] * al + ps;
        mi[i][r] = mall;
#pragma unroll
        for (int t = 0; t < 4; ++t) o[i][t][r] *= al;
      }

    // --- P: C-layout -> LDS -> A-layout (per-wave buffer, DS in-order) ---
#pragma unroll
    for (int i = 0; i < 2; ++i)
#pragma unroll
      for (int kt = 0; kt < 2; ++kt)
#pragma unroll
        for (int r = 0; r < 8; ++r) {
          unsigned M = i * 16u + r + 8u * half;
          pb[M * 32u + kt * 16u + mlane] = f2bf(s[i][kt][r]);
        }
    union { v16bf v; unsigned u[8]; } pa[2];
    const unsigned* p32 = (const unsigned*)pb;
#pragma unroll
    for (int i = 0; i < 2; ++i)
#pragma unroll
      for (int v = 0; v < 8; ++v)
        pa[i].u[v] = p32[((i * 16u + mlane) * 32u + a_kk(v, half)) >> 1];

    // --- O += P @ V; V B-fragments shared by both row sub-tiles ---
#pragma unroll
    for (int t = 0; t < 4; ++t) {
      unsigned d = t * 16u + mlane;
      const uint4* vp = (const uint4*)(vbase + (size_t)d * SEQ + jb + 16u * half);
      union { v16bf v; uint4 q[2]; } bv;
      bv.q[0] = vp[0]; bv.q[1] = vp[1];
#pragma unroll
      for (int i = 0; i < 2; ++i)
        o[i][t] = __builtin_amdgcn_wmma_f32_16x16x32_bf16(
            false, pa[i].v, false, bv.v, (short)0, o[i][t], false, false);
    }
  }

  // --- normalize and write attention output [B*T, C] bf16 ---
#pragma unroll
  for (int i = 0; i < 2; ++i)
#pragma unroll
    for (int r = 0; r < 8; ++r) {
      float inv = 1.0f / li[i][r];
      unsigned gm = b * SEQ + q0 + i * 16u + r + 8u * half;
#pragma unroll
      for (int t = 0; t < 4; ++t)
        attb[(size_t)gm * EMBED + h * HD + t * 16u + mlane] =
            f2bf(o[i][t][r] * inv);
    }
}

// ---------------------------------------------------------------------------
// Kernel 4: output projection. out = att @ out_w^T + out_b (f32 output)
// ---------------------------------------------------------------------------
__global__ __launch_bounds__(256) void proj_gemm(
    const unsigned short* __restrict__ ab,   // [4096,1024] bf16
    const unsigned short* __restrict__ wb,   // [1024,1024] bf16
    const float* __restrict__ bias,          // [1024]
    float* __restrict__ out) {               // [4096,1024] f32
  const unsigned lane  = threadIdx.x & 31u;
  const unsigned wid   = threadIdx.x >> 5;
  const unsigned mlane = lane & 15u;
  const unsigned half  = lane >> 4;
  const unsigned m0 = (blockIdx.x * 8u + wid) * 32u;
  const unsigned n0 = blockIdx.y * 64u;

  v8f acc[2][4] = {{{}, {}, {}, {}}, {{}, {}, {}, {}}};
  const unsigned* a32 = (const unsigned*)ab;

  for (unsigned k0 = 0; k0 < EMBED; k0 += 32) {
    union { v16bf v; unsigned u[8]; } a[2];
#pragma unroll
    for (int i = 0; i < 2; ++i)
#pragma unroll
      for (int v = 0; v < 8; ++v) {
        unsigned kk = a_kk(v, half) + k0;
        a[i].u[v] = a32[((m0 + i * 16u + mlane) * EMBED + kk) >> 1];
      }
#pragma unroll
    for (int t = 0; t < 4; ++t) {
      unsigned n = n0 + t * 16u + mlane;
      const uint4* bp = (const uint4*)(wb + n * EMBED + k0 + 16u * half);
      union { v16bf v; uint4 q[2]; } bm;
      bm.q[0] = bp[0]; bm.q[1] = bp[1];
#pragma unroll
      for (int i = 0; i < 2; ++i)
        acc[i][t] = __builtin_amdgcn_wmma_f32_16x16x32_bf16(
            false, a[i].v, false, bm.v, (short)0, acc[i][t], false, false);
    }
  }

#pragma unroll
  for (int t = 0; t < 4; ++t) {
    unsigned n = n0 + t * 16u + mlane;
    float bv = bias[n];
#pragma unroll
    for (int i = 0; i < 2; ++i)
#pragma unroll
      for (int r = 0; r < 8; ++r) {
        unsigned gm = m0 + i * 16u + r + 8u * half;
        out[(size_t)gm * EMBED + n] = acc[i][t][r] + bv;
      }
  }
}

// ---------------------------------------------------------------------------
// Launch
// ---------------------------------------------------------------------------
extern "C" void kernel_launch(void* const* d_in, const int* in_sizes, int n_in,
                              void* d_out, int out_size, void* d_ws, size_t ws_size,
                              hipStream_t stream) {
  const float* x      = (const float*)d_in[0];   // [2,2048,1024]
  const float* qkv_w  = (const float*)d_in[1];   // [3072,1024]
  const float* qkv_b  = (const float*)d_in[2];   // [3072]
  const float* out_w  = (const float*)d_in[3];   // [1024,1024]
  const float* out_b  = (const float*)d_in[4];   // [1024]
  float* out = (float*)d_out;

  const int nx  = NROW * EMBED;        // 4,194,304
  const int nw1 = 3 * EMBED * EMBED;   // 3,145,728
  const int nw2 = EMBED * EMBED;       // 1,048,576

  char* ws = (char*)d_ws;
  unsigned short* xb   = (unsigned short*)(ws);
  unsigned short* w1b  = (unsigned short*)(ws + 8388608);
  unsigned short* w2b  = (unsigned short*)(ws + 14680064);
  unsigned short* qb   = (unsigned short*)(ws + 16777216);
  unsigned short* kb   = (unsigned short*)(ws + 25165824);
  unsigned short* vtb  = (unsigned short*)(ws + 33554432);
  unsigned short* attb = (unsigned short*)(ws + 41943040);  // ends at 48 MB

  int total = nx + nw1 + nw2;
  cvt_kernel<<<(total + 255) / 256, 256, 0, stream>>>(
      x, qkv_w, out_w, xb, w1b, w2b, nx, nw1, nw2);

  qkv_gemm<<<dim3(NROW / 256, (3 * EMBED) / 64), 256, 0, stream>>>(
      xb, w1b, qkv_b, qb, kb, vtb);

  attn_kernel<<<(2 * NH * (SEQ / 32)) / 8, 256, 0, stream>>>(qb, kb, vtb, attb);

  proj_gemm<<<dim3(NROW / 256, EMBED / 64), 256, 0, stream>>>(
      attb, w2b, out_b, out);
}
